// Attention_86663850099018
// MI455X (gfx1250) — compile-verified, matrix operand order checked by reading, not run on hardware
//
#include <hip/hip_runtime.h>

// ---------------------------------------------------------------------------
// Fused cosine-similarity attention for MI455X (gfx1250, wave32, WMMA).
//   ref:  xn = x / ||x||_row ;  S = softmax(xn xn^T) ;  g = H * S h ; g/||g||
//   (the H factor cancels in the global-norm division, so it is dropped)
// Shapes: N=2048, B=8, E=256, H=512.  x:(N,B,E) f32, h:(N,B,H) f32.
//
// Data movement: K chunks (16 KB / 32 keys) are shared by all 8 waves of a
// workgroup through a double-buffered LDS tile filled with CDNA5 async loads
// (global_load_async_to_lds_b128, ASYNCcnt), one barrier per chunk. The Q
// tile is staged through LDS once and then held in registers (loop-invariant
// WMMA fragments). V slices are wave-private and stream from L2 with
// global_load_b128 clauses.
// ---------------------------------------------------------------------------

typedef __bf16 bf16_t;
typedef __attribute__((ext_vector_type(8)))  float  v8f;
typedef __attribute__((ext_vector_type(16))) bf16_t v16bf;
typedef __attribute__((ext_vector_type(8)))  bf16_t v8bf;

constexpr int N = 2048;
constexpr int B = 8;
constexpr int E = 256;
constexpr int H = 512;

constexpr int QT      = 16;          // queries per block
constexpr int WAVES   = 8;           // waves per block (256 threads)
constexpr int HW      = H / WAVES;   // 64 h-columns per wave
constexpr int HT      = HW / 16;     // 4 h-column tiles per wave
constexpr int KC      = N / 32;      // 64 key chunks of 32
constexpr int KSTRIDE = E + 16;      // 272 elems = 544 B: bank-rotating stride

// 16x32 bf16 fragment in the WMMA A-operand layout (== per-lane layout of the
// B operand of the transposed matrix):
//   lanes 0-15 : row = lane,    K = k0+0..7  and k0+16..23
//   lanes 16-31: row = lane-16, K = k0+8..15 and k0+24..31
__device__ __forceinline__ v16bf load_frag(const bf16_t* __restrict__ base,
                                           int stride, int k0, int lane) {
  const bf16_t* p = base + (size_t)(lane & 15) * stride + k0 + ((lane >> 4) << 3);
  v8bf lo = *(const v8bf*)(p);
  v8bf hi = *(const v8bf*)(p + 16);
  return __builtin_shufflevector(lo, hi, 0, 1, 2, 3, 4, 5, 6, 7,
                                         8, 9, 10, 11, 12, 13, 14, 15);
}

__device__ __forceinline__ void async_b128(uint32_t lds, const bf16_t* g) {
  asm volatile("global_load_async_to_lds_b128 %0, %1, off"
               :: "v"(lds), "v"(g) : "memory");
}

// Cooperative async copy of one 32x256 bf16 K chunk (16 KB) into padded LDS.
// 256 threads x 4 x b128: each thread owns 64 contiguous bytes of one row.
__device__ __forceinline__ void k_chunk_async(const bf16_t* __restrict__ gk,
                                              bf16_t* lk, int tid) {
  const int row  = tid >> 3;          // 0..31
  const int cole = (tid & 7) * 32;    // element offset of this thread's 64 B
  const bf16_t* g = gk + (size_t)row * E + cole;
  const uint32_t l = (uint32_t)(uintptr_t)(lk + (size_t)row * KSTRIDE + cole);
#pragma unroll
  for (int j = 0; j < 4; ++j) async_b128(l + j * 16, g + (size_t)j * 8);
}

// One-time async copy of the 16x256 bf16 Q tile (8 KB) into padded LDS.
__device__ __forceinline__ void q_tile_async(const bf16_t* __restrict__ gq,
                                             bf16_t* lq, int tid) {
  const int row  = tid >> 4;          // 0..15
  const int cole = (tid & 15) * 16;   // element offset of this thread's 32 B
  const bf16_t* g = gq + (size_t)row * E + cole;
  const uint32_t l = (uint32_t)(uintptr_t)(lq + (size_t)row * KSTRIDE + cole);
#pragma unroll
  for (int j = 0; j < 2; ++j) async_b128(l + j * 16, g + (size_t)j * 8);
}

// --------------------------- preprocessing --------------------------------

__global__ void zero_kernel(float* sumsq) {
  if (threadIdx.x == 0 && blockIdx.x == 0) *sumsq = 0.0f;
}

// Row-normalize x and convert to bf16, re-laid-out as (B, N, E).
__global__ __launch_bounds__(128) void prep_x_kernel(const float* __restrict__ x,
                                                     bf16_t* __restrict__ xn) {
  const int lane = threadIdx.x & 31;
  const int wid  = blockIdx.x * 4 + (threadIdx.x >> 5);
  if (wid >= N * B) return;                       // wave-uniform
  const int n = wid / B, b = wid % B;
  const float* row = x + ((size_t)n * B + b) * E; // (N,B,E): E contiguous
  float ss = 0.f;
#pragma unroll
  for (int e = lane; e < E; e += 32) { float v = row[e]; ss += v * v; }
#pragma unroll
  for (int m = 16; m; m >>= 1) ss += __shfl_xor(ss, m, 32);
  const float inv = rsqrtf(ss);
  bf16_t* orow = xn + ((size_t)b * N + n) * E;
#pragma unroll
  for (int e = lane; e < E; e += 32) orow[e] = (bf16_t)(row[e] * inv);
}

// Transpose h (N,B,H) f32 -> hT (B,H,N) bf16 (output-coalesced).
__global__ void prep_h_kernel(const float* __restrict__ h,
                              bf16_t* __restrict__ hT) {
  size_t i = (size_t)blockIdx.x * blockDim.x + threadIdx.x;  // over B*H*N
  if (i >= (size_t)B * H * N) return;
  const int n  = (int)(i % N);
  const size_t t = i / N;
  const int hc = (int)(t % H);
  const int b  = (int)(t / H);
  hT[i] = (bf16_t)h[((size_t)n * B + b) * H + hc];
}

// ----------------------------- main kernel --------------------------------

__global__ __launch_bounds__(256) void attn_kernel(const bf16_t* __restrict__ xn,
                                                   const bf16_t* __restrict__ hT,
                                                   float* __restrict__ out,
                                                   float* __restrict__ sumsq) {
  __shared__ bf16_t kbuf[2][32 * KSTRIDE];        // double-buffered K chunk
  __shared__ bf16_t qbuf[16 * KSTRIDE];           // Q staging tile

  const int tid   = threadIdx.x;
  const int lane  = tid & 31;
  const int wave  = tid >> 5;
  const int qt    = blockIdx.x % (N / QT);
  const int b     = blockIdx.x / (N / QT);
  const int q0    = qt * QT;
  const int hc0   = wave * HW;
  const int rlo   = lane & 15;
  const int khalf = (lane >> 4) << 3;             // 0 or 8

  const bf16_t* Qb = xn + (size_t)b * N * E;      // (N,E) normalized rows
  const bf16_t* Vt = hT + (size_t)b * H * N;      // (H,N)

  // Online-softmax state: each lane carries query (lane & 15).
  float m_run = -3.0e38f;
  float l_run = 0.0f;
  v8f acc[HT];
#pragma unroll
  for (int t = 0; t < HT; ++t) acc[t] = v8f{};

  // Prologue: async-fill Q tile and K chunk 0, then pin the Q-tile WMMA
  // fragments in registers (loop-invariant, 64 VGPRs).
  q_tile_async(Qb + (size_t)q0 * E, qbuf, tid);
  k_chunk_async(Qb, &kbuf[0][0], tid);
  asm volatile("s_wait_asynccnt 0" ::: "memory");
  __syncthreads();

  v16bf qf[E / 32];
#pragma unroll
  for (int e = 0; e < E / 32; ++e)
    qf[e] = load_frag(qbuf, KSTRIDE, e * 32, lane);

  for (int kc = 0; kc < KC; ++kc) {
    const int cur = kc & 1;
    // Invariant at loop top: kbuf[cur] is filled and visible to all waves.
    if (kc + 1 < KC)
      k_chunk_async(Qb + (size_t)(kc + 1) * 32 * E, &kbuf[cur ^ 1][0], tid);

    const bf16_t* kb = &kbuf[cur][0];

    // ---- S^T = K_tile x Q_tile^T, two 16-key subtiles, f32 acc ----
    v8f s0 = v8f{}, s1 = v8f{};
#pragma unroll
    for (int e = 0; e < E / 32; ++e) {
      v16bf kf0 = load_frag(kb,                KSTRIDE, e * 32, lane);
      v16bf kf1 = load_frag(kb + 16 * KSTRIDE, KSTRIDE, e * 32, lane);
      s0 = __builtin_amdgcn_wmma_f32_16x16x32_bf16(false, kf0, false, qf[e],
                                                   (short)0, s0, false, false);
      s1 = __builtin_amdgcn_wmma_f32_16x16x32_bf16(false, kf1, false, qf[e],
                                                   (short)0, s1, false, false);
    }

    // ---- online softmax: lane-local over 16 VGPRs + one xor-16 combine ----
    float tm = s0[0];
#pragma unroll
    for (int i = 1; i < 8; ++i) tm = fmaxf(tm, s0[i]);
#pragma unroll
    for (int i = 0; i < 8; ++i) tm = fmaxf(tm, s1[i]);
    tm = fmaxf(tm, __shfl_xor(tm, 16, 32));
    const float m_new = fmaxf(m_run, tm);
    const float corr  = __expf(m_run - m_new);

    // exp + pack into the P A-fragment (already in A layout: lane = query,
    // keys split 0-7/16-23 vs 8-15/24-31 exactly as the two subtiles landed).
    float rs = 0.0f;
    v16bf pf;
#pragma unroll
    for (int i = 0; i < 8; ++i) {
      float p = __expf(s0[i] - m_new); rs += p; pf[i] = (bf16_t)p;
    }
#pragma unroll
    for (int i = 0; i < 8; ++i) {
      float p = __expf(s1[i] - m_new); rs += p; pf[8 + i] = (bf16_t)p;
    }
    rs += __shfl_xor(rs, 16, 32);
    l_run = l_run * corr + rs;
    m_run = m_new;

    // Broadcast per-query rescale factors to accumulator rows (ds_bpermute).
    float bs[8];
#pragma unroll
    for (int v = 0; v < 8; ++v) bs[v] = __shfl(corr, v + khalf, 32);

    // ---- O += P (16x32) x V (32x16) per h-column tile (V from global) ----
#pragma unroll
    for (int t = 0; t < HT; ++t) {
      v8f a = acc[t];
#pragma unroll
      for (int v = 0; v < 8; ++v) a[v] *= bs[v];
      v16bf vf = load_frag(Vt + (size_t)(hc0 + t * 16) * N, N, kc * 32, lane);
      acc[t] = __builtin_amdgcn_wmma_f32_16x16x32_bf16(false, pf, false, vf,
                                                       (short)0, a, false, false);
    }

    // Complete the overlapped fill of the next buffer; the single barrier
    // both publishes it and retires all reads of the current buffer.
    if (kc + 1 < KC)
      asm volatile("s_wait_asynccnt 0" ::: "memory");
    __syncthreads();
  }

  // ---- finalize: divide by softmax denominator, write out, sum squares ----
  float bl[8];
#pragma unroll
  for (int v = 0; v < 8; ++v) bl[v] = 1.0f / __shfl(l_run, v + khalf, 32);

  const int rowbase = q0 + khalf;   // D layout: lanes 16-31 hold rows M+8
  float ss = 0.0f;
#pragma unroll
  for (int t = 0; t < HT; ++t) {
#pragma unroll
    for (int v = 0; v < 8; ++v) {
      const float o = acc[t][v] * bl[v];
      ss += o * o;
      out[((size_t)(rowbase + v) * B + b) * H + hc0 + t * 16 + rlo] = o;
    }
  }
#pragma unroll
  for (int m = 16; m; m >>= 1) ss += __shfl_xor(ss, m, 32);
  if (lane == 0) atomicAdd(sumsq, ss);
}

// -------------------------- global-norm scale -----------------------------

__global__ void scale_kernel(float* __restrict__ out, const float* __restrict__ sumsq,
                             size_t total) {
  const float inv = rsqrtf(*sumsq);
  for (size_t i = (size_t)blockIdx.x * blockDim.x + threadIdx.x; i < total;
       i += (size_t)gridDim.x * blockDim.x)
    out[i] *= inv;
}

// ------------------------------- launcher ---------------------------------

extern "C" void kernel_launch(void* const* d_in, const int* in_sizes, int n_in,
                              void* d_out, int out_size, void* d_ws, size_t ws_size,
                              hipStream_t stream) {
  const float* x = (const float*)d_in[0];   // (N,B,E) f32
  const float* h = (const float*)d_in[1];   // (N,B,H) f32
  float* out = (float*)d_out;               // (N,B,H) f32

  char* ws = (char*)d_ws;
  float*  sumsq = (float*)ws;                                            // 256 B
  bf16_t* xn = (bf16_t*)(ws + 256);                                      // B*N*E bf16
  bf16_t* hT = (bf16_t*)(ws + 256 + (size_t)B * N * E * sizeof(bf16_t)); // B*H*N bf16

  zero_kernel<<<1, 1, 0, stream>>>(sumsq);
  prep_x_kernel<<<(N * B + 3) / 4, 128, 0, stream>>>(x, xn);
  prep_h_kernel<<<(int)(((size_t)B * H * N + 255) / 256), 256, 0, stream>>>(h, hT);
  attn_kernel<<<B * (N / QT), WAVES * 32, 0, stream>>>(xn, hT, out, sumsq);
  scale_kernel<<<1024, 256, 0, stream>>>(out, sumsq, (size_t)N * B * H);
}
